// SAGEEMB_15444702397229
// MI455X (gfx1250) — matrix operand order repaired and checked by reading, not compiled, not run on hardware
//
#include <hip/hip_runtime.h>

// ---------------------------------------------------------------------------
// GraphSAGE 3-layer forward on MI455X (gfx1250, wave32, WMMA).
// B=8, N=4096, dims 128 -> 64 -> 32 -> 32.
// HBM-bound on adj (536MB fp32, streamed once per layer = 3 passes total,
// ~69us roofline at 23.3 TB/s). Binary adjacency converts to f16 losslessly
// in registers; activations are f16 with fp32 WMMA accumulation.
// B tiles are staged per-block into double-buffered LDS with gfx1250 async
// global->LDS loads (ASYNCcnt), cutting L2 read traffic 4x vs per-wave loads.
// ---------------------------------------------------------------------------

typedef __attribute__((ext_vector_type(16))) _Float16 v16h;
typedef __attribute__((ext_vector_type(8)))  _Float16 v8h;
typedef __attribute__((ext_vector_type(8)))  float    v8f;
typedef int sage_v4i __attribute__((vector_size(16)));

#define SAGE_AS1 __attribute__((address_space(1)))
#define SAGE_AS3 __attribute__((address_space(3)))

#define NB    8
#define NN    4096

#if defined(__AMDGCN__) && __has_builtin(__builtin_amdgcn_global_load_async_to_lds_b128) && __has_builtin(__builtin_amdgcn_s_wait_asynccnt)
#define SAGE_ASYNC_LDS 1
#else
#define SAGE_ASYNC_LDS 0
#endif

// ---------------------------------------------------------------------------
// Kernel 1: h fp32 [b][node][d]  ->  hT f16 [b][feat][node]
// (so the WMMA B operand is contiguous along K = node index)
// ---------------------------------------------------------------------------
__global__ __launch_bounds__(256) void sage_cvt_transpose_kernel(
    const float* __restrict__ h, _Float16* __restrict__ hT, int d)
{
    size_t i    = (size_t)blockIdx.x * blockDim.x + threadIdx.x;   // B*N*d total
    int    f    = (int)(i % d);
    size_t nd   = i / d;         // b*N + node
    int    node = (int)(nd & (NN - 1));
    int    b    = (int)(nd >> 12);
    hT[((size_t)(b * d + f)) * NN + node] = (_Float16)h[i];
}

// ---------------------------------------------------------------------------
// Kernel 2: t = adj @ h  (per batch [4096x4096] x [4096 x D], fp32 accum).
// Block = 4 waves, same batch b; wave owns 16 M-rows and ALL NT = D/16
// column tiles (adj streamed once; each A fragment feeds NT WMMAs).
// B tile (D feats x 32 nodes, f16) is identical for all 4 waves -> staged
// once per block in double-buffered LDS via async global->LDS loads.
// Lane layouts per CDNA5 ISA:
//   A 16x32 f16 : row = lane&15; elems e: K = (e>=8?16:0) + (lane>=16?8:0) + (e&7)
//   B 32x16 f16 : col = lane&15; K = (e>=8?16:0) + (lane>=16?8:0) + (e&7), same as A
//   C 16x16 f32 : col = lane&15; row = r + 8*(lane>=16) for element r
// WDEG: fuse in-degree reduction (exact: fp32 sums of 0/1) into this pass.
// ---------------------------------------------------------------------------
template<int NT, bool WDEG>
__global__ __launch_bounds__(128) void sage_gemm_adj_kernel(
    const float* __restrict__ adj, const _Float16* __restrict__ hT,
    float* __restrict__ t, float* __restrict__ deg_inv)
{
    constexpr int D    = NT * 16;
    constexpr int ROWB = 40;          // halfs per LDS row: 32 data + 8 pad (80B)
    constexpr int CPT  = D / 32;      // 16B staging chunks per thread (128 thr)

    __shared__ _Float16 bt[2][D * ROWB];

    const int tid  = threadIdx.x;
    const int lane = tid & 31;
    const int wave = tid >> 5;
    const int tile = blockIdx.x * 4 + wave;      // b*256 + mi
    const int mi   = tile & 255;
    const int b    = tile >> 8;
    const int m    = lane & 15;
    const int half = lane >> 4;

    // A: adjacency row (fp32); this lane covers K groups +0..7 and +16..23,
    // shifted by 8 for the upper half-wave.
    const float* arow = adj + ((size_t)b * NN + (size_t)mi * 16 + m) * NN + half * 8;
    // B source: transposed f16 activations for this batch
    const _Float16* hTb = hT + (size_t)b * D * NN;

    // cooperative B-tile staging: chunk ch covers feat f = ch>>2, 16B part ch&3
    auto stage = [&](int buf, int k0) {
        #pragma unroll
        for (int i = 0; i < CPT; ++i) {
            const int ch   = tid + i * 128;
            const int f    = ch >> 2;
            const int part = ch & 3;
#if SAGE_ASYNC_LDS
            __builtin_amdgcn_global_load_async_to_lds_b128(
                (SAGE_AS1 sage_v4i*)(hTb + (size_t)f * NN + k0 + part * 8),
                (SAGE_AS3 sage_v4i*)(&bt[buf][f * ROWB + part * 8]),
                0, 0);
#else
            *(sage_v4i*)(&bt[buf][f * ROWB + part * 8]) =
                *(const sage_v4i*)(hTb + (size_t)f * NN + k0 + part * 8);
#endif
        }
    };

    v8f c[NT] = {};
    float s = 0.0f;                              // row-sum partial (degree)

    stage(0, 0);
#if SAGE_ASYNC_LDS
    __builtin_amdgcn_s_wait_asynccnt(0);
#endif
    __syncthreads();

    int buf = 0;
    for (int k0 = 0; k0 < NN; k0 += 32) {
        if (k0 + 32 < NN) stage(buf ^ 1, k0 + 32);   // prefetch next B tile

        const float4* ap = (const float4*)(arow + k0);
        float4 f0 = ap[0];
        float4 f1 = ap[1];
        float4 f2 = ap[4];
        float4 f3 = ap[5];
        __builtin_prefetch(arow + k0 + 64, 0, 1);

        if (WDEG) {
            s += f0.x + f0.y + f0.z + f0.w;
            s += f1.x + f1.y + f1.z + f1.w;
            s += f2.x + f2.y + f2.z + f2.w;
            s += f3.x + f3.y + f3.z + f3.w;
        }

        v16h a;
        a[0]  = (_Float16)f0.x; a[1]  = (_Float16)f0.y;
        a[2]  = (_Float16)f0.z; a[3]  = (_Float16)f0.w;
        a[4]  = (_Float16)f1.x; a[5]  = (_Float16)f1.y;
        a[6]  = (_Float16)f1.z; a[7]  = (_Float16)f1.w;
        a[8]  = (_Float16)f2.x; a[9]  = (_Float16)f2.y;
        a[10] = (_Float16)f2.z; a[11] = (_Float16)f2.w;
        a[12] = (_Float16)f3.x; a[13] = (_Float16)f3.y;
        a[14] = (_Float16)f3.z; a[15] = (_Float16)f3.w;

        #pragma unroll
        for (int nt = 0; nt < NT; ++nt) {
            // conflict-free: dword start = (nt*16+m)*20 + half*4 (distinct mod 64)
            const v8h* bp = (const v8h*)(&bt[buf][(nt * 16 + m) * ROWB + half * 8]);
            v8h blo = bp[0];
            v8h bhi = bp[2];
            v16h bb;
            #pragma unroll
            for (int e = 0; e < 8; ++e) { bb[e] = blo[e]; bb[8 + e] = bhi[e]; }
            c[nt] = __builtin_amdgcn_wmma_f32_16x16x32_f16(
                        false, a, false, bb, (short)0, c[nt], false, false);
        }

#if SAGE_ASYNC_LDS
        __builtin_amdgcn_s_wait_asynccnt(0);     // next tile landed (this wave)
#endif
        __syncthreads();                         // all waves: loads done, reads done
        buf ^= 1;
    }

    #pragma unroll
    for (int nt = 0; nt < NT; ++nt) {
        float* trow = t + ((size_t)b * NN + (size_t)mi * 16 + half * 8) * D
                        + nt * 16 + m;
        #pragma unroll
        for (int r = 0; r < 8; ++r) trow[(size_t)r * D] = c[nt][r];
    }

    if (WDEG) {
        float stot = s + __shfl_xor(s, 16, 32);  // combine the two K-halves
        if (half == 0)
            deg_inv[(size_t)b * NN + mi * 16 + m] = 1.0f / fmaxf(stot, 1.0f);
    }
}

// ---------------------------------------------------------------------------
// Kernel 3: out = relu(h @ Ws + deg_inv * (adj@h) @ Wn + bias)
// One block per node; dout threads. h/t rows staged into LDS via gfx1250
// async global->LDS loads (fallback: plain loads). din <= 2*dout always.
// ---------------------------------------------------------------------------
__global__ void sage_combine_kernel(
    const float* __restrict__ h, const float* __restrict__ t,
    const float* __restrict__ deg_inv,
    const float* __restrict__ Ws, const float* __restrict__ Wn,
    const float* __restrict__ bias, float* __restrict__ out,
    int din, int dout)
{
    __shared__ float sh[128];
    __shared__ float st[128];
    const int node = blockIdx.x;          // b*N + node
    const int tid  = threadIdx.x;

#if SAGE_ASYNC_LDS
    const int nch = din >> 2;             // 16B chunks per row
    if (tid < nch) {
        __builtin_amdgcn_global_load_async_to_lds_b128(
            (SAGE_AS1 sage_v4i*)(h + (size_t)node * din + tid * 4),
            (SAGE_AS3 sage_v4i*)(&sh[tid * 4]),
            0, 0);
    } else if (tid < 2 * nch) {
        const int c2 = tid - nch;
        __builtin_amdgcn_global_load_async_to_lds_b128(
            (SAGE_AS1 sage_v4i*)(t + (size_t)node * din + c2 * 4),
            (SAGE_AS3 sage_v4i*)(&st[c2 * 4]),
            0, 0);
    }
    __builtin_amdgcn_s_wait_asynccnt(0);
    __syncthreads();
#else
    for (int k = tid; k < din; k += blockDim.x) {
        sh[k] = h[(size_t)node * din + k];
        st[k] = t[(size_t)node * din + k];
    }
    __syncthreads();
#endif

    const float dinv = deg_inv[node];
    float acc  = bias[tid];
    float accn = 0.0f;
    for (int k = 0; k < din; ++k) {
        acc  = fmaf(sh[k], Ws[k * dout + tid], acc);
        accn = fmaf(st[k], Wn[k * dout + tid], accn);
    }
    acc += dinv * accn;
    out[(size_t)node * dout + tid] = fmaxf(acc, 0.0f);
}

// ---------------------------------------------------------------------------
// Host orchestration
// ---------------------------------------------------------------------------
template<int NT, bool WDEG>
static void run_layer(const float* adj, const float* h_in, float* h_out,
                      _Float16* hT, float* tbuf, float* deg_inv,
                      const float* Ws, const float* Wn, const float* bias,
                      int dout, hipStream_t stream)
{
    constexpr int D = NT * 16;
    // 1) transpose + f16 convert activations
    {
        size_t elems = (size_t)NB * NN * D;
        sage_cvt_transpose_kernel<<<(unsigned)(elems / 256), 256, 0, stream>>>(
            h_in, hT, D);
    }
    // 2) aggregation GEMM via WMMA (adj streamed once; deg fused if WDEG)
    sage_gemm_adj_kernel<NT, WDEG><<<(NB * 256) / 4, 128, 0, stream>>>(
        adj, hT, tbuf, deg_inv);
    // 3) fused epilogue
    sage_combine_kernel<<<NB * NN, dout, 0, stream>>>(
        h_in, tbuf, deg_inv, Ws, Wn, bias, h_out, D, dout);
}

extern "C" void kernel_launch(void* const* d_in, const int* in_sizes, int n_in,
                              void* d_out, int out_size, void* d_ws, size_t ws_size,
                              hipStream_t stream)
{
    const float* x   = (const float*)d_in[0];
    const float* adj = (const float*)d_in[1];
    const float* Ws0 = (const float*)d_in[2];
    const float* Wn0 = (const float*)d_in[3];
    const float* b0  = (const float*)d_in[4];
    const float* Ws1 = (const float*)d_in[5];
    const float* Wn1 = (const float*)d_in[6];
    const float* b1  = (const float*)d_in[7];
    const float* Ws2 = (const float*)d_in[8];
    const float* Wn2 = (const float*)d_in[9];
    const float* b2  = (const float*)d_in[10];
    float* out = (float*)d_out;

    char* ws = (char*)d_ws;
    // workspace layout (all 256B-aligned offsets), ~38 MB total
    float*     deg_inv = (float*)ws;                                 // 128 KB
    _Float16*  hT      = (_Float16*)(ws + (128u << 10));             //   8 MB
    float*     tbuf    = (float*)(ws + (128u << 10) + (8u << 20));   //  16 MB
    float*     h1      = (float*)(ws + (128u << 10) + (24u << 20));  //   8 MB
    float*     h2      = (float*)(ws + (128u << 10) + (32u << 20));  //   4 MB

    // layer 0: 128 -> 64 (degree reduction fused into the adj stream)
    run_layer<8, true >(adj, x,  h1,  hT, tbuf, deg_inv, Ws0, Wn0, b0, 64, stream);
    // layer 1: 64 -> 32
    run_layer<4, false>(adj, h1, h2,  hT, tbuf, deg_inv, Ws1, Wn1, b1, 32, stream);
    // layer 2: 32 -> 32 (writes final embeddings)
    run_layer<2, false>(adj, h2, out, hT, tbuf, deg_inv, Ws2, Wn2, b2, 32, stream);
}
